// Adapter_7370163880020
// MI455X (gfx1250) — matrix operand order repaired and checked by reading, not compiled, not run on hardware
//
#include <hip/hip_runtime.h>

// ---------------------------------------------------------------------------
// MI455X (gfx1250, wave32) implementation of the bidirectional cross-attention
// adapter.  All GEMMs run on v_wmma_f32_16x16x32_bf16; softmax statistics and
// accumulation stay in f32.  Score rows are materialized per-block in LDS
// (16 x SLEN x 4B <= 128KB, fits CDNA5's 320KB/WGP LDS).  The Q tile is staged
// into LDS once per block via the Tensor Data Mover (tensor_load_to_lds).
// ---------------------------------------------------------------------------

typedef __attribute__((ext_vector_type(16))) __bf16 v16bf;
typedef __attribute__((ext_vector_type(8)))  __bf16 v8bf;
typedef __attribute__((ext_vector_type(8)))  float  v8f;
typedef __attribute__((ext_vector_type(4)))  unsigned int u32x4;
typedef __attribute__((ext_vector_type(8)))  unsigned int u32x8;

#define B_    16
#define LS_   2048
#define LL_   1024
#define CIN_  256
#define COUT_ 1024

__device__ __forceinline__ v8f zero8() {
  v8f z = {0.f, 0.f, 0.f, 0.f, 0.f, 0.f, 0.f, 0.f};
  return z;
}

__device__ __forceinline__ v8f wmma_bf16(v16bf a, v16bf b, v8f c) {
  // D = A(16x32 bf16) x B(32x16 bf16) + C(16x16 f32)
  return __builtin_amdgcn_wmma_f32_16x16x32_bf16(
      /*neg_a=*/false, a, /*neg_b=*/false, b,
      /*c_mod=*/(short)0, c, /*reuse_a=*/false, /*reuse_b=*/false);
}

// ---------------------------------------------------------------------------
// Kernel 1: transpose adapter weight W[f32, CIN x COUT] -> WT[bf16, COUT x CIN]
// so WMMA B-tile loads are contiguous (n = lane&15 fixed, K contiguous).
// ---------------------------------------------------------------------------
__global__ void prep_w_kernel(const float* __restrict__ W,
                              __bf16* __restrict__ WT) {
  const int idx = blockIdx.x * 256 + threadIdx.x;  // over COUT_*CIN_
  const int n = idx >> 8;   // cout
  const int k = idx & 255;  // cin
  WT[idx] = (__bf16)W[(size_t)k * COUT_ + n];
}

// ---------------------------------------------------------------------------
// Kernel 2: language prep.  One block per row of [B*LL, 1024]:
//   - copy f32 row into the output slot (return value #2)
//   - bf16 copy (row-major) and bf16 transposed copy [b][d][LL]
//   - rinv_l[row] = 1/||row||
// ---------------------------------------------------------------------------
__global__ void prep_lang_kernel(const float* __restrict__ lf,
                                 float* __restrict__ out_copy,
                                 __bf16* __restrict__ lB,
                                 __bf16* __restrict__ lBT,
                                 float* __restrict__ rinv) {
  const int row = blockIdx.x;  // 0 .. B*LL-1
  const int b = row >> 10;     // LL = 1024
  const int l = row & 1023;
  const int tid = threadIdx.x;  // 128 threads
  const float* src = lf + (size_t)row * COUT_;
  float sum = 0.f;
  for (int c = tid; c < COUT_; c += 128) {
    const float v = src[c];
    out_copy[(size_t)row * COUT_ + c] = v;
    const __bf16 h = (__bf16)v;
    lB[(size_t)row * COUT_ + c] = h;
    lBT[((size_t)b * COUT_ + c) * LL_ + l] = h;
    sum += v * v;
  }
  for (int off = 16; off > 0; off >>= 1) sum += __shfl_xor(sum, off, 32);
  __shared__ float r4[4];
  if ((tid & 31) == 0) r4[tid >> 5] = sum;
  __syncthreads();
  if (tid == 0) rinv[row] = rsqrtf(r4[0] + r4[1] + r4[2] + r4[3] + 1e-20f);
}

// ---------------------------------------------------------------------------
// Kernel 3: channel adapter GEMM.
//   speech[r, 0:1024] = speech_feature[r, 0:256] @ W + bias
// Block = 256 threads (8 waves); each wave computes one 16x16 tile of a
// 16-row x 128-col slab.  K = 256 -> 8 WMMA per tile.  Writes f32 output
// (return value #4), bf16 copy, bf16 transposed copy.
// ---------------------------------------------------------------------------
__global__ void adapter_kernel(const float* __restrict__ sf,
                               const __bf16* __restrict__ WT,
                               const float* __restrict__ bias,
                               float* __restrict__ out_s,
                               __bf16* __restrict__ sB,
                               __bf16* __restrict__ sBT) {
  const int tid = threadIdx.x;
  const int lane = tid & 31;
  const int wave = tid >> 5;                       // 0..7
  const int r_base = blockIdx.x * 16;              // global speech row
  const int n_base = blockIdx.y * 128 + wave * 16; // output column tile
  const int nl = lane & 15;
  const int hi = lane >> 4;

  v8f acc = zero8();
  const float* arow = sf + ((size_t)(r_base + nl)) * CIN_ + (hi ? 16 : 0);
  const __bf16* brow = WT + (size_t)(n_base + nl) * CIN_ + hi * 16;
#pragma unroll
  for (int kt = 0; kt < CIN_ / 32; ++kt) {
    const float* pa = arow + kt * 32;
    v8f f0 = *(const v8f*)pa;
    v8f f1 = *(const v8f*)(pa + 8);
    v16bf a;
#pragma unroll
    for (int j = 0; j < 8; ++j) {
      a[j] = (__bf16)f0[j];
      a[j + 8] = (__bf16)f1[j];
    }
    v16bf bm = *(const v16bf*)(brow + kt * 32);
    acc = wmma_bf16(a, bm, acc);
  }

  const int n_gl = n_base + nl;
  const float bb = bias[n_gl];
#pragma unroll
  for (int i = 0; i < 8; ++i) {
    const int r = r_base + i + 8 * hi;  // C/D layout: VGPR i -> row i+8*hi
    const float v = acc[i] + bb;
    const size_t idx = (size_t)r * COUT_ + n_gl;
    out_s[idx] = v;
    const __bf16 h = (__bf16)v;
    sB[idx] = h;
    const int bb_ = r >> 11;   // LS = 2048
    const int s = r & 2047;
    sBT[((size_t)bb_ * COUT_ + n_gl) * LS_ + s] = h;
  }
}

// ---------------------------------------------------------------------------
// Kernel 4: per-row reciprocal L2 norm of the f32 adapter output.
// ---------------------------------------------------------------------------
__global__ void norm_rows_kernel(const float* __restrict__ src,
                                 float* __restrict__ rinv) {
  const size_t row = blockIdx.x;
  const int tid = threadIdx.x;  // 128
  const float* p = src + row * COUT_;
  float sum = 0.f;
  for (int c = tid; c < COUT_; c += 128) {
    const float v = p[c];
    sum += v * v;
  }
  for (int off = 16; off > 0; off >>= 1) sum += __shfl_xor(sum, off, 32);
  __shared__ float r4[4];
  if ((tid & 31) == 0) r4[tid >> 5] = sum;
  __syncthreads();
  if (tid == 0) rinv[row] = rsqrtf(r4[0] + r4[1] + r4[2] + r4[3] + 1e-20f);
}

// ---------------------------------------------------------------------------
// Kernel 5: one-direction cross attention (used for both l->s and s->l).
//   out[16 rows of Q] = softmax( (Q K^T) * rq * rk ) @ V
// Q : bf16 [B*Lq, 1024] (unnormalized), K : bf16 [B*SLEN, 1024],
// VT: bf16 [B][1024][SLEN] (V transposed so B-tiles are contiguous).
// Per block: 256 threads (8 waves).  The 16x1024 bf16 Q tile is DMA'd into
// LDS once via the Tensor Data Mover; the full score row S[16][SLEN] lives in
// f32 LDS; P (bf16) in LDS; then P @ V with each wave owning a 128-wide
// d-slice (8 f32 accumulator tiles = 64 VGPRs).
// ---------------------------------------------------------------------------
__global__ void attn_kernel(const __bf16* __restrict__ Q,
                            const __bf16* __restrict__ K,
                            const __bf16* __restrict__ VT,
                            const float* __restrict__ rq,
                            const float* __restrict__ rk,
                            float* __restrict__ out, int Lq, int SLEN) {
  extern __shared__ char smem[];
  float* S = (float*)smem;                                   // 16*SLEN f32
  __bf16* P = (__bf16*)(smem + (size_t)16 * SLEN * 4);       // 16*SLEN bf16
  __bf16* Qs = (__bf16*)(smem + (size_t)16 * SLEN * 6);      // 16*1024 bf16
  __shared__ float rowmax[16];
  __shared__ float rowinv[16];
  __shared__ float qr[16];
  __shared__ float red[256];

  const int tid = threadIdx.x;
  const int lane = tid & 31;
  const int wave = tid >> 5;       // 0..7
  const int r_base = blockIdx.x * 16;  // global Q row (b*Lq + q)
  const int b = r_base / Lq;
  const int nl = lane & 15;
  const int hi = lane >> 4;

  const __bf16* Qbase = Q + (size_t)r_base * COUT_;

  // ---- Stage Q tile (16 x 1024 bf16 = 32KB) into LDS with the TDM ----
  if (wave == 0) {
    const unsigned ldsQoff =
        __builtin_amdgcn_groupstaticsize() + (unsigned)(16 * SLEN * 6);
    const unsigned long long ga = (unsigned long long)(const void*)Qbase;
    u32x4 g0;
    g0[0] = 1u;                                  // count=1 (valid user D#)
    g0[1] = ldsQoff;                             // lds_addr (bytes)
    g0[2] = (unsigned)(ga & 0xffffffffull);      // global_addr[31:0]
    g0[3] = (unsigned)((ga >> 32) & 0x1ffffffull) | (2u << 30);  // type=2
    u32x8 g1;
    g1[0] = (1u << 16);      // workgroup_mask=0, data_size=1 (2 bytes)
    g1[1] = (1024u << 16);   // atomic_barrier_addr=0 | tensor_dim0[15:0]=1024
    g1[2] = (16u << 16);     // tensor_dim0[31:16]=0 | tensor_dim1[15:0]=16
    g1[3] = (1024u << 16);   // tensor_dim1[31:16]=0 | tile_dim0=1024
    g1[4] = 16u;             // tile_dim1=16, tile_dim2=0
    g1[5] = 1024u;           // tensor_dim0_stride[31:0]=1024 elements
    g1[6] = 0u;              // stride hi, tensor_dim1_stride lo
    g1[7] = 0u;              // tensor_dim1_stride hi
    asm volatile("tensor_load_to_lds %0, %1"
                 :
                 : "s"(g0), "s"(g1)
                 : "memory");
    __builtin_amdgcn_s_wait_tensorcnt(0);
  }
  if (tid < 16) qr[tid] = rq[r_base + tid];
  __syncthreads();

  // ---- Phase 1: S = (Q K^T) * rq[m] * rk[n] ----
  const int ntiles = SLEN >> 4;
  const __bf16* arow = Qs + (size_t)nl * COUT_ + (hi ? 16 : 0);  // LDS A rows
  for (int tt = wave; tt < ntiles; tt += 8) {
    v8f acc = zero8();
    const __bf16* brow =
        K + ((size_t)b * SLEN + tt * 16 + nl) * COUT_ + hi * 16;
    if (tt + 8 < ntiles) {  // prefetch next K tile row for this lane
      __builtin_prefetch(
          K + ((size_t)b * SLEN + (tt + 8) * 16 + nl) * COUT_ + hi * 16, 0, 0);
    }
    for (int kt = 0; kt < COUT_ / 32; ++kt) {
      const __bf16* pa = arow + kt * 32;
      v8bf c0 = *(const v8bf*)pa;        // ds loads
      v8bf c1 = *(const v8bf*)(pa + 8);
      v16bf a = __builtin_shufflevector(c0, c1, 0, 1, 2, 3, 4, 5, 6, 7, 8, 9,
                                        10, 11, 12, 13, 14, 15);
      v16bf bm = *(const v16bf*)(brow + kt * 32);
      acc = wmma_bf16(a, bm, acc);
    }
    const float rs = rk[(size_t)b * SLEN + tt * 16 + nl];
#pragma unroll
    for (int i = 0; i < 8; ++i) {
      const int m = i + 8 * hi;
      S[m * SLEN + tt * 16 + nl] = acc[i] * qr[m] * rs;
    }
  }
  __syncthreads();

  // ---- Phase 2a: row softmax (max, sum of exp) and P = softmax(S) in bf16 --
  {
    const int row = tid >> 4;
    const int col0 = tid & 15;
    float m = -3.4e38f;
    for (int s = col0; s < SLEN; s += 16) m = fmaxf(m, S[row * SLEN + s]);
    red[tid] = m;
    __syncthreads();
    if (tid < 16) {
      float mm = -3.4e38f;
      for (int j = 0; j < 16; ++j) mm = fmaxf(mm, red[tid * 16 + j]);
      rowmax[tid] = mm;
    }
    __syncthreads();
    const float mrow = rowmax[row];
    float sum = 0.f;
    for (int s = col0; s < SLEN; s += 16) sum += __expf(S[row * SLEN + s] - mrow);
    red[tid] = sum;
    __syncthreads();
    if (tid < 16) {
      float ss = 0.f;
      for (int j = 0; j < 16; ++j) ss += red[tid * 16 + j];
      rowinv[tid] = 1.0f / ss;
    }
    __syncthreads();
    for (int rr = 0; rr < 16; ++rr) {
      const float mx = rowmax[rr];
      const float iv = rowinv[rr];
      for (int s = tid; s < SLEN; s += 256)
        P[rr * SLEN + s] = (__bf16)(__expf(S[rr * SLEN + s] - mx) * iv);
    }
    __syncthreads();
  }

  // ---- Phase 2b: out = P @ V ;  wave owns d-slice [wave*128, wave*128+128) --
  v8f o[8];
#pragma unroll
  for (int t = 0; t < 8; ++t) o[t] = zero8();
  const int kTiles = SLEN >> 5;
  const __bf16* prow = P + (size_t)nl * SLEN + (hi ? 16 : 0);
  for (int kt = 0; kt < kTiles; ++kt) {
    const __bf16* pa = prow + kt * 32;
    v8bf c0 = *(const v8bf*)pa;        // LDS reads
    v8bf c1 = *(const v8bf*)(pa + 8);
    v16bf a = __builtin_shufflevector(c0, c1, 0, 1, 2, 3, 4, 5, 6, 7, 8, 9, 10,
                                      11, 12, 13, 14, 15);
#pragma unroll
    for (int t = 0; t < 8; ++t) {
      const int n_gl = wave * 128 + t * 16 + nl;
      const __bf16* pb =
          VT + ((size_t)b * COUT_ + n_gl) * SLEN + kt * 32 + hi * 16;
      v16bf bm = *(const v16bf*)pb;
      o[t] = wmma_bf16(a, bm, o[t]);
    }
  }
#pragma unroll
  for (int t = 0; t < 8; ++t) {
    const int n_gl = wave * 128 + t * 16 + nl;
#pragma unroll
    for (int i = 0; i < 8; ++i) {
      const int m = i + 8 * hi;
      out[((size_t)r_base + m) * COUT_ + n_gl] = o[t][i];
    }
  }
}

// ---------------------------------------------------------------------------
// Host-side launch.  Workspace layout (bytes):
//   sB16  bf16 [B][LS][1024]   @ 0          (64 MB)
//   sB16T bf16 [B][1024][LS]   @ 64  MB     (64 MB)
//   lB16  bf16 [B][LL][1024]   @ 128 MB     (32 MB)
//   lB16T bf16 [B][1024][LL]   @ 160 MB     (32 MB)
//   WT    bf16 [1024][256]     @ 192 MB     (512 KB)
//   rinv_s f32 [B*LS]          @ +512 KB    (128 KB)
//   rinv_l f32 [B*LL]          @ +128 KB    (64 KB)
// Output layout (f32, concatenated): lang_pred | lang_copy | speech_pred | speech
// ---------------------------------------------------------------------------
extern "C" void kernel_launch(void* const* d_in, const int* in_sizes, int n_in,
                              void* d_out, int out_size, void* d_ws,
                              size_t ws_size, hipStream_t stream) {
  const float* sf = (const float*)d_in[0];    // [16,2048,256]
  const float* lf = (const float*)d_in[1];    // [16,1024,1024]
  const float* W = (const float*)d_in[2];     // [256,1024]
  const float* bias = (const float*)d_in[3];  // [1024]

  float* out = (float*)d_out;
  float* out_lpred = out;                                  // 16,777,216
  float* out_lcopy = out + (size_t)16777216;               // 16,777,216
  float* out_spred = out + (size_t)33554432;               // 33,554,432
  float* out_sp = out + (size_t)67108864;                  // speech (adapter)

  char* ws = (char*)d_ws;
  __bf16* sB = (__bf16*)(ws);
  __bf16* sBT = (__bf16*)(ws + (size_t)67108864);
  __bf16* lB = (__bf16*)(ws + (size_t)134217728);
  __bf16* lBT = (__bf16*)(ws + (size_t)167772160);
  __bf16* WT = (__bf16*)(ws + (size_t)201326592);
  float* rinv_s = (float*)(ws + (size_t)201326592 + 524288);
  float* rinv_l = (float*)(ws + (size_t)201326592 + 524288 + 131072);

  // 1) W transpose -> bf16
  prep_w_kernel<<<(COUT_ * CIN_) / 256, 256, 0, stream>>>(W, WT);
  // 2) language: copy/bf16/bf16^T/norm
  prep_lang_kernel<<<B_ * LL_, 128, 0, stream>>>(lf, out_lcopy, lB, lBT,
                                                 rinv_l);
  // 3) adapter GEMM (WMMA bf16): speech f32 + bf16 + bf16^T
  adapter_kernel<<<dim3((B_ * LS_) / 16, COUT_ / 128), 256, 0, stream>>>(
      sf, WT, bias, out_sp, sB, sBT);
  // 4) speech row norms
  norm_rows_kernel<<<B_ * LS_, 128, 0, stream>>>(out_sp, rinv_s);
  // 5) language -> speech attention (softmax over LS), out = P @ speech
  attn_kernel<<<(B_ * LL_) / 16, 256, (size_t)16 * LS_ * 6 + 32768, stream>>>(
      lB, sB, sBT, rinv_l, rinv_s, out_lpred, LL_, LS_);
  // 6) speech -> language attention (softmax over LL), out = P @ language
  attn_kernel<<<(B_ * LS_) / 16, 256, (size_t)16 * LL_ * 6 + 32768, stream>>>(
      sB, lB, lBT, rinv_s, rinv_l, out_spred, LS_, LL_);
}
